// DynamicKoopmanOperator_32109175505770
// MI455X (gfx1250) — matrix operand order repaired and checked by reading, not compiled, not run on hardware
//
#include <hip/hip_runtime.h>
#include <math.h>

#define KDT 0.01f

typedef __attribute__((ext_vector_type(16))) _Float16 v16h;
typedef __attribute__((ext_vector_type(8)))  _Float16 v8h;
typedef __attribute__((ext_vector_type(8)))  float    v8f;
typedef __attribute__((ext_vector_type(4)))  float    v4f;
typedef __attribute__((ext_vector_type(2)))  float    v2f;

union V16H { v16h v; v8h h[2]; };

// ---------- prep: f32 -> f16 conversions ----------
__global__ void cvt_transpose_f16(const float* __restrict__ W,
                                  _Float16* __restrict__ Wt, int K) {
  int idx = blockIdx.x * blockDim.x + threadIdx.x;
  if (idx < K * K) {
    int n = idx / K, k = idx - n * K;        // Wt[n][k] = W[k][n]
    Wt[idx] = (_Float16)W[(size_t)k * K + n];
  }
}

__global__ void cvt_f16(const float* __restrict__ X,
                        _Float16* __restrict__ Xh, int n) {
  int idx = blockIdx.x * blockDim.x + threadIdx.x;
  if (idx < n) Xh[idx] = (_Float16)X[idx];
}

// ---------- WMMA 16x16 tile core: C(16x16,f32) = A[M,K] x Bt[N,K]^T ----------
__device__ __forceinline__ v8f wmma_tile_f16(const _Float16* __restrict__ A,
                                             const _Float16* __restrict__ Bt,
                                             int K, int m0, int n0) {
  int lane = threadIdx.x & 31;
  int g    = lane >> 4;   // lane half-group
  int rc   = lane & 15;   // A-row / B-col within tile
  const _Float16* arow = A  + (size_t)(m0 + rc) * K;
  const _Float16* bcol = Bt + (size_t)(n0 + rc) * K;
  v8f acc = {};
  #pragma unroll
  for (int k0 = 0; k0 < 256; k0 += 32) {
    V16H a;
    // 16-bit A 16x32 layout: halves 0..7 -> K=k0+g*8+0..7 ; halves 8..15 -> K=k0+16+g*8+0..7
    a.h[0] = *(const v8h*)(arow + k0 + g * 8);
    a.h[1] = *(const v8h*)(arow + k0 + 16 + g * 8);
    // 16-bit B 32x16 layout: lane holds col rc, halves 0..7 -> K=k0+g*8+0..7 ; halves 8..15 -> K=k0+16+g*8+0..7
    V16H bb;
    bb.h[0] = *(const v8h*)(bcol + k0 + g * 8);
    bb.h[1] = *(const v8h*)(bcol + k0 + 16 + g * 8);
    v16h b = bb.v;
    acc = __builtin_amdgcn_wmma_f32_16x16x32_f16(false, a.v, false, b,
                                                 (short)0, acc, false, false);
  }
  return acc;
}

// ---------- GEMM1: H = tanh(X @ W1 + b1), stored as f16 ----------
__global__ void gemm1_tanh(const _Float16* __restrict__ Xh,
                           const _Float16* __restrict__ W1t,
                           const float* __restrict__ b1,
                           _Float16* __restrict__ H, int Bsz, int K) {
  int wave = threadIdx.x >> 5;
  int tile = blockIdx.x * (blockDim.x >> 5) + wave;
  int ntiles = K >> 4;
  int mt = tile / ntiles, nt = tile - mt * ntiles;
  if (mt >= (Bsz >> 4)) return;              // wave-uniform
  int m0 = mt << 4, n0 = nt << 4;
  v8f acc = wmma_tile_f16(Xh, W1t, K, m0, n0);
  int lane = threadIdx.x & 31;
  int g = lane >> 4, rc = lane & 15;
  int n = n0 + rc;
  float bias = b1[n];
  #pragma unroll
  for (int r = 0; r < 8; ++r) {
    int m = m0 + r + 8 * g;                  // C layout: VGPR r -> M=r (+8 for hi lanes)
    H[(size_t)m * K + n] = (_Float16)tanhf(acc[r] + bias);
  }
}

// ---------- GEMM2: eigs = H @ W2 + b2 -> Koopman coefs (a,b) per pair ----------
__global__ void gemm2_coef(const _Float16* __restrict__ H,
                           const _Float16* __restrict__ W2t,
                           const float* __restrict__ b2,
                           v2f* __restrict__ coef, int Bsz, int K) {
  int wave = threadIdx.x >> 5;
  int tile = blockIdx.x * (blockDim.x >> 5) + wave;
  int ntiles = K >> 4;
  int mt = tile / ntiles, nt = tile - mt * ntiles;
  if (mt >= (Bsz >> 4)) return;
  int m0 = mt << 4, n0 = nt << 4;
  v8f acc = wmma_tile_f16(H, W2t, K, m0, n0);
  int lane = threadIdx.x & 31;
  int g = lane >> 4, rc = lane & 15;
  int n = n0 + rc;
  float bias = b2[n];
  int half = K >> 1;
  #pragma unroll
  for (int r = 0; r < 8; ++r) {
    float val   = acc[r] + bias;
    float other = __shfl_xor(val, 1, 32);    // pair even (mu) with odd (omega) column
    float mu    = (rc & 1) ? other : val;
    float omega = (rc & 1) ? val   : other;
    float e = expf(mu * KDT);
    float s, c;
    sincosf(omega * KDT, &s, &c);
    if (!(rc & 1)) {
      int m = m0 + r + 8 * g;
      v2f ab; ab.x = e * c; ab.y = e * s;
      coef[(size_t)m * half + (n >> 1)] = ab;
    }
  }
}

// ---------- rollout: 256-step 2x2 recurrence, NT float4 streaming stores ----------
__global__ void rollout(const float* __restrict__ X,
                        const v2f* __restrict__ coef,
                        float* __restrict__ out, int Bsz, int K, int T) {
  int b = blockIdx.x * (blockDim.x >> 6) + (threadIdx.x >> 6);
  if (b >= Bsz) return;
  int j = threadIdx.x & 63;                  // owns k = 4j .. 4j+3 (two eigen-pairs)
  v4f z = *(const v4f*)(X + (size_t)b * K + 4 * j);
  int half = K >> 1;
  v2f c0 = coef[(size_t)b * half + 2 * j];
  v2f c1 = coef[(size_t)b * half + 2 * j + 1];
  float* p = out + (size_t)b * T * K + 4 * j;
  for (int t = 0; t < T; ++t) {
    float z0 = c0.x * z.x - c0.y * z.y;
    float z1 = c0.y * z.x + c0.x * z.y;
    float z2 = c1.x * z.z - c1.y * z.w;
    float z3 = c1.y * z.z + c1.x * z.w;
    z.x = z0; z.y = z1; z.z = z2; z.w = z3;
    __builtin_nontemporal_store(z, (v4f*)p); // 268MB stream: bypass temporal caching
    p += K;
  }
}

extern "C" void kernel_launch(void* const* d_in, const int* in_sizes, int n_in,
                              void* d_out, int out_size, void* d_ws, size_t ws_size,
                              hipStream_t stream) {
  const float* x  = (const float*)d_in[0];
  const float* W1 = (const float*)d_in[1];
  const float* b1 = (const float*)d_in[2];
  const float* W2 = (const float*)d_in[3];
  const float* b2 = (const float*)d_in[4];
  const int K   = in_sizes[2];               // 256
  const int Bsz = in_sizes[0] / K;           // 1024
  const int T   = out_size / (Bsz * K);      // 256
  float* out = (float*)d_out;

  // workspace carve-out (~2.3 MB total)
  char* ws = (char*)d_ws;
  size_t offs = 0;
  auto carve = [&](size_t bytes) {
    char* p = ws + offs;
    offs = (offs + bytes + 255) & ~(size_t)255;
    return p;
  };
  _Float16* W1t = (_Float16*)carve((size_t)K * K * 2);
  _Float16* W2t = (_Float16*)carve((size_t)K * K * 2);
  _Float16* Xh  = (_Float16*)carve((size_t)Bsz * K * 2);
  _Float16* Hh  = (_Float16*)carve((size_t)Bsz * K * 2);
  v2f*      cf  = (v2f*)carve((size_t)Bsz * (K / 2) * sizeof(v2f));

  const int thr = 256;
  cvt_transpose_f16<<<(K * K + thr - 1) / thr, thr, 0, stream>>>(W1, W1t, K);
  cvt_transpose_f16<<<(K * K + thr - 1) / thr, thr, 0, stream>>>(W2, W2t, K);
  cvt_f16<<<(Bsz * K + thr - 1) / thr, thr, 0, stream>>>(x, Xh, Bsz * K);

  int tiles = (Bsz / 16) * (K / 16);         // 1024 tiles, one wave each
  int blk = 128;                             // 4 waves per block
  int grid = (tiles + (blk / 32) - 1) / (blk / 32);
  gemm1_tanh<<<grid, blk, 0, stream>>>(Xh, W1t, b1, Hh, Bsz, K);
  gemm2_coef<<<grid, blk, 0, stream>>>(Hh, W2t, b2, cf, Bsz, K);

  rollout<<<(Bsz + 3) / 4, 256, 0, stream>>>(x, cf, out, Bsz, K, T);
}
